// VectorQuantizer_21998822490528
// MI455X (gfx1250) — matrix operand order repaired and checked by reading, not compiled, not run on hardware
//
#include <hip/hip_runtime.h>
#include <hip/hip_bf16.h>

typedef __attribute__((ext_vector_type(16))) _Float16 v16h;
typedef __attribute__((ext_vector_type(8)))  _Float16 v8h;
typedef __attribute__((ext_vector_type(8)))  float    v8f;

#define NUM_EMBED 1024
#define EMBED_DIM 256
#define HW        1024                  // 32*32
#define N_TOTAL   (64 * HW)             // 65536 rows
#define OUT_ELEMS (64 * EMBED_DIM * HW) // 16777216
#define TILE_M    64                    // rows per block (4 row-tiles of 16)
#define ZH_STR    264                   // padded f16 row stride (bank 4m, 16B aligned)
#define ZF_STR    257                   // padded f32 row stride (bank m+c)

// ---------------------------------------------------------------------------
// Kernel 0: zero the loss accumulator in workspace
// ---------------------------------------------------------------------------
__global__ void vq_init_kernel(float* __restrict__ acc) {
    if (threadIdx.x == 0) *acc = 0.0f;
}

// ---------------------------------------------------------------------------
// Kernel 1: fp32 codebook -> f16 (WMMA B operand) + fp32 ||e_n||^2 per code
// ---------------------------------------------------------------------------
__global__ __launch_bounds__(256) void vq_convert_kernel(
    const float* __restrict__ embed,
    _Float16* __restrict__ embed_h,
    float* __restrict__ norms) {
    const int n = blockIdx.x;
    const int c = threadIdx.x;
    float v = embed[(size_t)n * EMBED_DIM + c];
    embed_h[(size_t)n * EMBED_DIM + c] = (_Float16)v;
    __shared__ float red[256];
    red[c] = v * v;
    __syncthreads();
    for (int s = 128; s > 0; s >>= 1) {
        if (c < s) red[c] += red[c + s];
        __syncthreads();
    }
    if (c == 0) norms[n] = red[0];
}

// ---------------------------------------------------------------------------
// Kernel 2: main VQ kernel. 1024 blocks x 128 threads (4 waves).
// Block owns 64 rows (one b, 64 contiguous hw). Waves split the 1024 codes
// (256 each); each loaded B fragment feeds 4 WMMAs (one per 16-row tile).
// ---------------------------------------------------------------------------
__global__ __launch_bounds__(128, 1) void vq_main_kernel(
    const float*    __restrict__ z,
    const float*    __restrict__ embed,
    const _Float16* __restrict__ embed_h,
    const float*    __restrict__ norms,
    float*          __restrict__ out,
    float*          __restrict__ acc) {

    __shared__ __attribute__((aligned(16))) _Float16 zh[TILE_M * ZH_STR]; // ~33 KB
    __shared__ float zf[TILE_M * ZF_STR];                                 // ~64 KB
    __shared__ float nrm[NUM_EMBED];                                      // 4 KB
    __shared__ float redv[4][TILE_M];
    __shared__ int   redi[4][TILE_M];
    __shared__ int   fidx[TILE_M];
    __shared__ float lsum[128];

    const int tid  = threadIdx.x;
    const int lane = tid & 31;
    const int wave = tid >> 5;

    const int tile = blockIdx.x;       // 0..1023
    const int b    = tile >> 4;        // batch index
    const int hw0  = (tile & 15) << 6; // first hw of the 64-row tile

    // ---- stage z tile into LDS (fp32 for loss, f16 for WMMA A) ----
    {
        const int sm = tid & 63;   // row within tile -> contiguous hw
        const int sc = tid >> 6;   // 0..1
        for (int ci = 0; ci < 128; ++ci) {
            const int c = ci * 2 + sc;
            const float v = z[((size_t)(b * EMBED_DIM + c)) * HW + hw0 + sm];
            zf[sm * ZF_STR + c] = v;
            zh[sm * ZH_STR + c] = (_Float16)v;
        }
    }
    for (int i = tid; i < NUM_EMBED; i += 128) nrm[i] = norms[i];
    __syncthreads();

    // A layout: lane L holds row M=L%16; lanes 0-15 carry K 0-15,
    // lanes 16-31 carry K 16-31 (elems 0-7 = first 8, 8-15 = next 8).
    const int am = lane & 15;
    const int kb = (lane < 16) ? 0 : 16;
    // B layout: lane L holds column N=L%16; lanes 0-15 carry K 0-15,
    // lanes 16-31 carry K 16-31 -> one 32-byte contiguous load per K-tile.
    const int kb16 = (lane < 16) ? 0 : 16;

    float bestv[4][8];
    int   besti[4][8];
#pragma unroll
    for (int mt = 0; mt < 4; ++mt)
#pragma unroll
        for (int r = 0; r < 8; ++r) { bestv[mt][r] = 3.4e38f; besti[mt][r] = 0; }

    for (int nt = 0; nt < 16; ++nt) {
        const int n0 = wave * 256 + nt * 16;
        const int n  = n0 + (lane & 15);
        if (nt < 15) // prefetch next B tile (global_prefetch_b8)
            __builtin_prefetch(&embed_h[(size_t)(n + 16) * EMBED_DIM], 0, 3);

        v8f acc8[4] = {};
#pragma unroll
        for (int kt = 0; kt < 8; ++kt) {
            const v16h bfrag =
                *(const v16h*)&embed_h[(size_t)n * EMBED_DIM + kt * 32 + kb16];
#pragma unroll
            for (int mt = 0; mt < 4; ++mt) {
                const int rowb = (mt * 16 + am) * ZH_STR + kt * 32 + kb;
                const v8h lo = *(const v8h*)&zh[rowb];
                const v8h hi = *(const v8h*)&zh[rowb + 8];
                v16h a;
#pragma unroll
                for (int i = 0; i < 8; ++i) { a[i] = lo[i]; a[8 + i] = hi[i]; }
                acc8[mt] = __builtin_amdgcn_wmma_f32_16x16x32_f16(
                    false, a, false, bfrag, (short)0, acc8[mt], false, false);
            }
        }
        // distance score: ||e||^2 - 2 z.e  (||z||^2 is row-constant, drop it)
        const float en = nrm[n];
#pragma unroll
        for (int mt = 0; mt < 4; ++mt)
#pragma unroll
            for (int r = 0; r < 8; ++r) {
                const float d = en - 2.0f * acc8[mt][r];
                if (d < bestv[mt][r]) { bestv[mt][r] = d; besti[mt][r] = n; }
            }
    }

    // ---- cross-lane argmin within each 16-lane half (xor masks stay inside) ----
#pragma unroll
    for (int mt = 0; mt < 4; ++mt)
#pragma unroll
        for (int r = 0; r < 8; ++r) {
            float v = bestv[mt][r];
            int   i = besti[mt][r];
            for (int m = 8; m >= 1; m >>= 1) {
                const float ov = __shfl_xor(v, m, 32);
                const int   oi = __shfl_xor(i, m, 32);
                if (ov < v || (ov == v && oi < i)) { v = ov; i = oi; }
            }
            bestv[mt][r] = v; besti[mt][r] = i;
        }
    // D layout: lanes 0-15 hold rows 0-7 (reg r=row), lanes 16-31 hold rows 8-15
    if (lane == 0)
#pragma unroll
        for (int mt = 0; mt < 4; ++mt)
            for (int r = 0; r < 8; ++r) {
                redv[wave][mt * 16 + r] = bestv[mt][r];
                redi[wave][mt * 16 + r] = besti[mt][r];
            }
    if (lane == 16)
#pragma unroll
        for (int mt = 0; mt < 4; ++mt)
            for (int r = 0; r < 8; ++r) {
                redv[wave][mt * 16 + 8 + r] = bestv[mt][r];
                redi[wave][mt * 16 + 8 + r] = besti[mt][r];
            }
    __syncthreads();

    // ---- cross-wave argmin (waves covered disjoint 256-code ranges) ----
    if (tid < TILE_M) {
        float v = redv[0][tid];
        int   i = redi[0][tid];
        for (int w = 1; w < 4; ++w) {
            const float ov = redv[w][tid];
            const int   oi = redi[w][tid];
            if (ov < v || (ov == v && oi < i)) { v = ov; i = oi; }
        }
        fidx[tid] = i;
    }
    __syncthreads();

    // ---- gather fp32 codebook rows -> coalesced (B,C,H,W) stores + loss ----
    float ls = 0.0f;
    {
        const int m  = tid & 63;
        const int cg = tid >> 6;
        const int e  = fidx[m];
        for (int ci = 0; ci < 128; ++ci) {
            const int c  = ci * 2 + cg;
            const float q  = embed[(size_t)e * EMBED_DIM + c];
            const float zv = zf[m * ZF_STR + c];
            const float d  = zv - q;
            ls += d * d;
            out[((size_t)(b * EMBED_DIM + c)) * HW + hw0 + m] = q;
        }
    }
    lsum[tid] = ls;
    __syncthreads();
    for (int s = 64; s > 0; s >>= 1) {
        if (tid < s) lsum[tid] += lsum[tid + s];
        __syncthreads();
    }
    if (tid == 0) atomicAdd(acc, lsum[0]);
}

// ---------------------------------------------------------------------------
// Kernel 3: finalize loss: (0.25 + 1.0) * mean((z - q)^2)
// ---------------------------------------------------------------------------
__global__ void vq_final_kernel(const float* __restrict__ acc,
                                float* __restrict__ out_loss) {
    if (threadIdx.x == 0)
        *out_loss = *acc * (1.25f / (float)OUT_ELEMS);
}

// ---------------------------------------------------------------------------
extern "C" void kernel_launch(void* const* d_in, const int* in_sizes, int n_in,
                              void* d_out, int out_size, void* d_ws, size_t ws_size,
                              hipStream_t stream) {
    const float* z     = (const float*)d_in[0]; // (64, 256, 32, 32)
    const float* embed = (const float*)d_in[1]; // (1024, 256)
    float* out = (float*)d_out;                 // 16777216 quantized + 1 loss

    char* ws = (char*)d_ws;
    float*    acc     = (float*)ws;                                    // 4 B (padded)
    _Float16* embed_h = (_Float16*)(ws + 256);                         // 512 KB
    float*    norms   = (float*)(ws + 256 + (size_t)NUM_EMBED * EMBED_DIM * sizeof(_Float16));

    vq_init_kernel<<<1, 1, 0, stream>>>(acc);
    vq_convert_kernel<<<NUM_EMBED, 256, 0, stream>>>(embed, embed_h, norms);
    vq_main_kernel<<<N_TOTAL / TILE_M, 128, 0, stream>>>(z, embed, embed_h, norms, out, acc);
    vq_final_kernel<<<1, 1, 0, stream>>>(acc, out + OUT_ELEMS);
}
